// MultiHeadSelfAttention_2937757630951
// MI455X (gfx1250) — matrix operand order repaired and checked by reading, not compiled, not run on hardware
//
#include <hip/hip_runtime.h>
#include <hip/hip_bf16.h>

// ---------------- problem constants ----------------
#define T_SEQ   4096
#define C_MODEL 1024
#define N_HEADS 16
#define HEAD_D  64
#define N_QKV   (3 * C_MODEL)   // 3072

typedef __attribute__((ext_vector_type(16))) _Float16 v16h;
typedef __attribute__((ext_vector_type(8)))  _Float16 v8h;
typedef __attribute__((ext_vector_type(8)))  float    v8f;

static __device__ __forceinline__ v16h cat8(v8h a, v8h b) {
    return __builtin_shufflevector(a, b, 0,1,2,3,4,5,6,7,8,9,10,11,12,13,14,15);
}
static __device__ __forceinline__ v8f wmma_f16(v16h a, v16h b, v8f c) {
    return __builtin_amdgcn_wmma_f32_16x16x32_f16(false, a, false, b, (short)0, c,
                                                  false, false);
}
// A-fragment (16x32 f16, row-major src): lane holds row lo, k = hi*8 + {0..7,16..23}
static __device__ __forceinline__ v16h load_afrag(const _Float16* __restrict__ base) {
    v8h a0 = *(const v8h*)(base);
    v8h a1 = *(const v8h*)(base + 16);
    return cat8(a0, a1);
}
// B-fragment (32x16 f16) from B^T row-major: lane holds col lo, k = hi*8 + {0..7,16..23}
static __device__ __forceinline__ v16h load_bfrag(const _Float16* __restrict__ base) {
    v8h b0 = *(const v8h*)(base);
    v8h b1 = *(const v8h*)(base + 16);
    return cat8(b0, b1);
}

// ---------------- kernel 1: f32 -> f16 cast ----------------
__global__ __launch_bounds__(256) void cast_f32_f16(const float* __restrict__ src,
                                                    _Float16* __restrict__ dst, int n) {
    int i = blockIdx.x * 256 + threadIdx.x;
    if (i < n) dst[i] = (_Float16)src[i];
}

// ---------------- kernel 2: QKV GEMM (32x64 per wave) ----------------
// qkv = x(4096x1024) @ w_qkv^T(1024x3072); scatter into Qh[h][t][d], Kh[h][t][d], Vt[h][d][t]
__global__ __launch_bounds__(128) void qkv_gemm(const _Float16* __restrict__ xh,
                                                const _Float16* __restrict__ wh,
                                                _Float16* __restrict__ Qh,
                                                _Float16* __restrict__ Kh,
                                                _Float16* __restrict__ Vt) {
    const int lane = threadIdx.x & 31;
    const int wid  = threadIdx.x >> 5;
    const int wg   = blockIdx.x * 4 + wid;          // 6144 wave jobs
    const int mb   = wg / 48;                        // 128 blocks of 32 rows
    const int ng   = wg % 48;                        // 48 groups of 4 n-tiles
    const int hi   = lane >> 4;
    const int lo   = lane & 15;

    v8f acc[2][4];
    #pragma unroll
    for (int mi = 0; mi < 2; ++mi)
        #pragma unroll
        for (int i = 0; i < 4; ++i) acc[mi][i] = (v8f)(0.f);

    const _Float16* arow0 = xh + (size_t)(mb * 32 + lo) * C_MODEL + hi * 8;
    const _Float16* arow1 = arow0 + (size_t)16 * C_MODEL;
    for (int kk = 0; kk < C_MODEL / 32; ++kk) {
        v16h a0 = load_afrag(arow0 + kk * 32);
        v16h a1 = load_afrag(arow1 + kk * 32);
        #pragma unroll
        for (int nt = 0; nt < 4; ++nt) {
            int n = (ng * 4 + nt) * 16 + lo;
            v16h bfrag = load_bfrag(wh + (size_t)n * C_MODEL + kk * 32 + hi * 8);
            acc[0][nt] = wmma_f16(a0, bfrag, acc[0][nt]);
            acc[1][nt] = wmma_f16(a1, bfrag, acc[1][nt]);
        }
    }

    #pragma unroll
    for (int mi = 0; mi < 2; ++mi) {
        #pragma unroll
        for (int nt = 0; nt < 4; ++nt) {
            const int n      = (ng * 4 + nt) * 16 + lo;
            const int which  = n >> 10;        // 0=Q 1=K 2=V (uniform per nt)
            const int within = n & 1023;
            const int h      = within >> 6;    // uniform per nt
            const int d      = within & 63;
            #pragma unroll
            for (int r = 0; r < 8; ++r) {
                const int t = mb * 32 + mi * 16 + r + hi * 8;
                _Float16 v = (_Float16)acc[mi][nt][r];
                if (which == 0)      Qh[((size_t)h * T_SEQ + t) * HEAD_D + d] = v;
                else if (which == 1) Kh[((size_t)h * T_SEQ + t) * HEAD_D + d] = v;
                else                 Vt[((size_t)h * HEAD_D + d) * T_SEQ + t] = v;
            }
        }
    }
}

// ---------------- kernel 3: flash attention (32 queries per wave) ----------------
__global__ __launch_bounds__(128) void flash_attn(const _Float16* __restrict__ Qh,
                                                  const _Float16* __restrict__ Kh,
                                                  const _Float16* __restrict__ Vt,
                                                  _Float16* __restrict__ attnh) {
    __shared__ _Float16 lds[4][2 * 16 * 32];   // 2KB P staging per wave
    const int lane = threadIdx.x & 31;
    const int wid  = threadIdx.x >> 5;
    const int wg   = blockIdx.x * 4 + wid;           // 2048 wave jobs
    const int h    = wg >> 7;                        // 16 heads
    const int tb   = (wg & 127) * 32;                // 128 query blocks of 32
    const int hi   = lane >> 4;
    const int lo   = lane & 15;
    const float L2E = 1.44269504088896340736f;

    const _Float16* Q = Qh + (size_t)h * T_SEQ * HEAD_D;
    const _Float16* K = Kh + (size_t)h * T_SEQ * HEAD_D;
    const _Float16* V = Vt + (size_t)h * HEAD_D * T_SEQ;
    _Float16* myLds = lds[wid];

    // Q A-fragments: 2 query tiles x (d=0..31, d=32..63)
    v16h qa[2][2];
    #pragma unroll
    for (int mi = 0; mi < 2; ++mi) {
        const _Float16* qrow = Q + (size_t)(tb + mi * 16 + lo) * HEAD_D + hi * 8;
        qa[mi][0] = load_afrag(qrow);
        qa[mi][1] = load_afrag(qrow + 32);
    }

    v8f o[2][4];
    float m[2][8], l[2][8];
    #pragma unroll
    for (int mi = 0; mi < 2; ++mi) {
        #pragma unroll
        for (int i = 0; i < 4; ++i) o[mi][i] = (v8f)(0.f);
        #pragma unroll
        for (int r = 0; r < 8; ++r) { m[mi][r] = -1e30f; l[mi][r] = 0.f; }
    }

    for (int j = 0; j < T_SEQ / 32; ++j) {
        const int sb = j * 32;
        // ---- shared K B-fragments: 2 s-tiles x 2 k-steps ----
        const _Float16* kr0 = K + (size_t)(sb + lo) * HEAD_D + hi * 8;
        const _Float16* kr1 = K + (size_t)(sb + 16 + lo) * HEAD_D + hi * 8;
        v16h kb00 = load_bfrag(kr0), kb01 = load_bfrag(kr0 + 32);
        v16h kb10 = load_bfrag(kr1), kb11 = load_bfrag(kr1 + 32);
        v8f s[2][2];
        #pragma unroll
        for (int mi = 0; mi < 2; ++mi) {
            v8f t0 = (v8f)(0.f), t1 = (v8f)(0.f);
            t0 = wmma_f16(qa[mi][0], kb00, t0);
            t0 = wmma_f16(qa[mi][1], kb01, t0);
            t1 = wmma_f16(qa[mi][0], kb10, t1);
            t1 = wmma_f16(qa[mi][1], kb11, t1);
            s[mi][0] = t0; s[mi][1] = t1;
        }
        // ---- online softmax per query tile ----
        #pragma unroll
        for (int mi = 0; mi < 2; ++mi) {
            float mn[8];
            #pragma unroll
            for (int r = 0; r < 8; ++r) {
                s[mi][0][r] *= 0.125f;  s[mi][1][r] *= 0.125f;   // 1/sqrt(64)
                mn[r] = fmaxf(m[mi][r], fmaxf(s[mi][0][r], s[mi][1][r]));
            }
            #pragma unroll
            for (int mask = 1; mask <= 8; mask <<= 1) {
                #pragma unroll
                for (int r = 0; r < 8; ++r)
                    mn[r] = fmaxf(mn[r], __shfl_xor(mn[r], mask, 32));
            }
            float ps[8];
            #pragma unroll
            for (int r = 0; r < 8; ++r) {
                float alpha = __builtin_amdgcn_exp2f((m[mi][r] - mn[r]) * L2E);
                float p0 = __builtin_amdgcn_exp2f((s[mi][0][r] - mn[r]) * L2E);
                float p1 = __builtin_amdgcn_exp2f((s[mi][1][r] - mn[r]) * L2E);
                s[mi][0][r] = p0; s[mi][1][r] = p1; ps[r] = p0 + p1;
                m[mi][r] = mn[r];
                l[mi][r] *= alpha;
                o[mi][0][r] *= alpha; o[mi][1][r] *= alpha;
                o[mi][2][r] *= alpha; o[mi][3][r] *= alpha;
            }
            #pragma unroll
            for (int mask = 1; mask <= 8; mask <<= 1) {
                #pragma unroll
                for (int r = 0; r < 8; ++r)
                    ps[r] += __shfl_xor(ps[r], mask, 32);
            }
            #pragma unroll
            for (int r = 0; r < 8; ++r) l[mi][r] += ps[r];
            // P (C layout) -> LDS
            #pragma unroll
            for (int r = 0; r < 8; ++r) {
                const int row = r + hi * 8;
                myLds[mi * 512 + row * 32 + lo]      = (_Float16)s[mi][0][r];
                myLds[mi * 512 + row * 32 + 16 + lo] = (_Float16)s[mi][1][r];
            }
        }
        asm volatile("s_wait_dscnt 0" ::: "memory");
        v16h pa0 = load_afrag(myLds + lo * 32 + hi * 8);
        v16h pa1 = load_afrag(myLds + 512 + lo * 32 + hi * 8);
        // ---- O += P @ V : V fragments shared across both query tiles ----
        #pragma unroll
        for (int dt = 0; dt < 4; ++dt) {
            const _Float16* vb = V + (size_t)(dt * 16 + lo) * T_SEQ + sb + hi * 8;
            v16h vfrag = load_bfrag(vb);
            o[0][dt] = wmma_f16(pa0, vfrag, o[0][dt]);
            o[1][dt] = wmma_f16(pa1, vfrag, o[1][dt]);
        }
    }
    // ---- normalize, store f16 for the output projection ----
    #pragma unroll
    for (int mi = 0; mi < 2; ++mi) {
        float linv[8];
        #pragma unroll
        for (int r = 0; r < 8; ++r) linv[r] = 1.f / l[mi][r];
        #pragma unroll
        for (int dt = 0; dt < 4; ++dt) {
            #pragma unroll
            for (int r = 0; r < 8; ++r) {
                const int t = tb + mi * 16 + r + hi * 8;
                attnh[(size_t)t * C_MODEL + h * HEAD_D + dt * 16 + lo] =
                    (_Float16)(o[mi][dt][r] * linv[r]);
            }
        }
    }
}

// ---------------- kernel 4: output projection + bias (32x64 per wave) ----------------
__global__ __launch_bounds__(128) void out_gemm(const _Float16* __restrict__ ah,
                                                const _Float16* __restrict__ wh,
                                                const float* __restrict__ bias,
                                                float* __restrict__ out) {
    const int lane = threadIdx.x & 31;
    const int wid  = threadIdx.x >> 5;
    const int wg   = blockIdx.x * 4 + wid;   // 2048 wave jobs
    const int mb   = wg >> 4;                // 128 blocks of 32 rows
    const int ng   = wg & 15;                // 16 groups of 4 n-tiles
    const int hi   = lane >> 4;
    const int lo   = lane & 15;

    v8f acc[2][4];
    #pragma unroll
    for (int mi = 0; mi < 2; ++mi)
        #pragma unroll
        for (int i = 0; i < 4; ++i) acc[mi][i] = (v8f)(0.f);

    const _Float16* arow0 = ah + (size_t)(mb * 32 + lo) * C_MODEL + hi * 8;
    const _Float16* arow1 = arow0 + (size_t)16 * C_MODEL;
    for (int kk = 0; kk < C_MODEL / 32; ++kk) {
        v16h a0 = load_afrag(arow0 + kk * 32);
        v16h a1 = load_afrag(arow1 + kk * 32);
        #pragma unroll
        for (int nt = 0; nt < 4; ++nt) {
            int n = (ng * 4 + nt) * 16 + lo;
            v16h bfrag = load_bfrag(wh + (size_t)n * C_MODEL + kk * 32 + hi * 8);
            acc[0][nt] = wmma_f16(a0, bfrag, acc[0][nt]);
            acc[1][nt] = wmma_f16(a1, bfrag, acc[1][nt]);
        }
    }
    #pragma unroll
    for (int mi = 0; mi < 2; ++mi) {
        #pragma unroll
        for (int nt = 0; nt < 4; ++nt) {
            const int n = (ng * 4 + nt) * 16 + lo;
            const float bv = bias[n];
            #pragma unroll
            for (int r = 0; r < 8; ++r) {
                const int t = mb * 32 + mi * 16 + r + hi * 8;
                out[(size_t)t * C_MODEL + n] = acc[mi][nt][r] + bv;
            }
        }
    }
}

// ---------------- host launch ----------------
extern "C" void kernel_launch(void* const* d_in, const int* in_sizes, int n_in,
                              void* d_out, int out_size, void* d_ws, size_t ws_size,
                              hipStream_t stream) {
    const float* x     = (const float*)d_in[0];
    const float* wqkv  = (const float*)d_in[1];
    const float* wout  = (const float*)d_in[2];
    const float* bout  = (const float*)d_in[3];
    float* out = (float*)d_out;

    char* ws = (char*)d_ws;
    const size_t MB = 1024 * 1024;
    _Float16* xh    = (_Float16*)(ws + 0);        //  8 MB: x f16          [4096][1024]
    _Float16* wqh   = (_Float16*)(ws + 8  * MB);  //  6 MB: w_qkv f16      [3072][1024]
    _Float16* woh   = (_Float16*)(ws + 14 * MB);  //  2 MB: w_out f16      [1024][1024]
    _Float16* Qh    = (_Float16*)(ws + 16 * MB);  //  8 MB: Q  [16][4096][64]
    _Float16* Kh    = (_Float16*)(ws + 24 * MB);  //  8 MB: K  [16][4096][64]
    _Float16* Vt    = (_Float16*)(ws + 32 * MB);  //  8 MB: V^T [16][64][4096]
    _Float16* attnh = (_Float16*)(ws + 40 * MB);  //  8 MB: attn f16 [4096][1024]

    const int nx = T_SEQ * C_MODEL;      // 4M
    const int nq = N_QKV * C_MODEL;      // 3M
    const int nw = C_MODEL * C_MODEL;    // 1M
    cast_f32_f16<<<(nx + 255) / 256, 256, 0, stream>>>(x,    xh,  nx);
    cast_f32_f16<<<(nq + 255) / 256, 256, 0, stream>>>(wqkv, wqh, nq);
    cast_f32_f16<<<(nw + 255) / 256, 256, 0, stream>>>(wout, woh, nw);

    // 128 m-blocks * 48 n-groups = 6144 waves / 4 per block
    qkv_gemm<<<1536, 128, 0, stream>>>(xh, wqh, Qh, Kh, Vt);
    // 16 heads * 128 query blocks = 2048 waves / 4 per block
    flash_attn<<<512, 128, 0, stream>>>(Qh, Kh, Vt, attnh);
    // 128 m-blocks * 16 n-groups = 2048 waves / 4 per block
    out_gemm<<<512, 128, 0, stream>>>(attnh, woh, bout, out);
}
